// get_model_19387482374287
// MI455X (gfx1250) — matrix operand order repaired and checked by reading, not compiled, not run on hardware
//
#include <hip/hip_runtime.h>

// ---------------------------------------------------------------------------
// CDNA5 (gfx1250) implementation of the PointNet++-style set abstraction +
// channel attention + CSA block.  wave32, WMMA f32_16x16x32_f16 GEMMs with
// 16x64 register blocking (4 WMMAs per A fragment).
// ---------------------------------------------------------------------------

typedef __attribute__((ext_vector_type(16))) _Float16 v16h;
typedef __attribute__((ext_vector_type(8)))  _Float16 v8h;
typedef __attribute__((ext_vector_type(8)))  float    v8f;

#define NB   8
#define NPTS 4096
#define NS   1024
#define NK   32
#define BSK  262144LL   // NB*NS*NK
#define BS   8192       // NB*NS

// ---------------------------------------------------------------------------
// 1) Farthest point sampling: one workgroup per batch, xyz staged in LDS,
//    per-thread distance state in registers, block arg-max each of 1024 steps.
// ---------------------------------------------------------------------------
__global__ __launch_bounds__(256) void k_fps(const float* __restrict__ xyz,
                                             float* __restrict__ new_xyz,
                                             int* __restrict__ fps_idx)
{
    __shared__ float sx[NPTS], sy[NPTS], sz[NPTS];
    __shared__ float rv[256];
    __shared__ int   ri[256];
    __shared__ int   s_far;
    const int b = blockIdx.x, t = threadIdx.x;
    const float* px = xyz + (long long)b * 3 * NPTS;
    for (int n = t; n < NPTS; n += 256) {
        sx[n] = px[n]; sy[n] = px[NPTS + n]; sz[n] = px[2 * NPTS + n];
    }
    float dloc[16];
#pragma unroll
    for (int j = 0; j < 16; ++j) dloc[j] = 1e10f;
    if (t == 0) s_far = 0;
    __syncthreads();

    for (int s = 0; s < NS; ++s) {
        const int far = s_far;
        const float cx = sx[far], cy = sy[far], cz = sz[far];
        if (t == 0) {
            fps_idx[b * NS + s] = far;
            float* nz = new_xyz + ((long long)b * NS + s) * 3;
            nz[0] = cx; nz[1] = cy; nz[2] = cz;
        }
        float bv = -1.0f; int bi = 0;
#pragma unroll
        for (int j = 0; j < 16; ++j) {
            const int n = t + j * 256;
            const float dx = sx[n] - cx, dy = sy[n] - cy, dz = sz[n] - cz;
            const float d  = dx * dx + dy * dy + dz * dz;
            const float dm = fminf(dloc[j], d);
            dloc[j] = dm;
            if (dm > bv) { bv = dm; bi = n; }   // strict > : first (lowest) index wins
        }
        rv[t] = bv; ri[t] = bi;
        __syncthreads();
        for (int off = 128; off > 0; off >>= 1) {
            if (t < off) {
                const float v2 = rv[t + off]; const int i2 = ri[t + off];
                if (v2 > rv[t] || (v2 == rv[t] && i2 < ri[t])) { rv[t] = v2; ri[t] = i2; }
            }
            __syncthreads();
        }
        if (t == 0) s_far = ri[0];
        __syncthreads();
    }
}

// ---------------------------------------------------------------------------
// 2) KNN (top-32 smallest distances, iterative min with per-lane bitmask) +
//    gather + concat + pad, written directly as f16 rows [BSK, 96] in the
//    exact row-major layout the WMMA GEMM consumes.
//    One wave32 per query; 8 queries (waves) per block; xyz in LDS.
// ---------------------------------------------------------------------------
__global__ __launch_bounds__(256) void k_knn_gather(
    const float* __restrict__ xyz, const float* __restrict__ csa_f,
    const float* __restrict__ csa_p, const float* __restrict__ new_xyz,
    _Float16* __restrict__ Apts, _Float16* __restrict__ Axyz)
{
    __shared__ float sx[NPTS], sy[NPTS], sz[NPTS];
    const int t = threadIdx.x;
    const int lane = t & 31, wv = t >> 5;
    const int b = blockIdx.x >> 7;                    // 128 blocks per batch
    const int s = ((blockIdx.x & 127) << 3) + wv;     // 8 queries per block
    const float* px = xyz + (long long)b * 3 * NPTS;
    for (int n = t; n < NPTS; n += 256) {
        sx[n] = px[n]; sy[n] = px[NPTS + n]; sz[n] = px[2 * NPTS + n];
    }
    __syncthreads();

    const float* q = new_xyz + ((long long)b * NS + s) * 3;
    const float qx = q[0], qy = q[1], qz = q[2];

    unsigned used[4] = {0u, 0u, 0u, 0u};   // candidate j used-mask (j = n>>5, n%32==lane)
    int myidx = 0;                          // neighbor index owned by lane k==lane
    for (int k = 0; k < NK; ++k) {
        float bv = 3.4e38f; int bi = 0x7fffffff;
        for (int j = 0; j < 128; ++j) {
            if ((used[j >> 5] >> (j & 31)) & 1u) continue;
            const int n = (j << 5) | lane;
            const float dx = sx[n] - qx, dy = sy[n] - qy, dz = sz[n] - qz;
            const float d = dx * dx + dy * dy + dz * dz;
            if (d < bv || (d == bv && n < bi)) { bv = d; bi = n; }
        }
#pragma unroll
        for (int off = 16; off > 0; off >>= 1) {     // wave32 min-reduce (ds_permute path)
            const float ov = __shfl_xor(bv, off, 32);
            const int   oi = __shfl_xor(bi, off, 32);
            if (ov < bv || (ov == bv && oi < bi)) { bv = ov; bi = oi; }
        }
        if ((bi & 31) == lane) { const int j = bi >> 5; used[j >> 5] |= 1u << (j & 31); }
        if (k == lane) myidx = bi;
    }

    // gather phase: lane handles neighbor k = lane
    const float gx = sx[myidx] - qx, gy = sy[myidx] - qy, gz = sz[myidx] - qz;
    const long long row = (((long long)b << 10) + s) * NK + lane;
    _Float16* ap = Apts + row * 96;
    _Float16* ax = Axyz + row * 96;
    const float* fsrc = csa_f + (long long)b * 64 * NPTS + myidx;
    const float* psrc = csa_p + (long long)b * 64 * NPTS + myidx;
#pragma unroll 8
    for (int d = 0; d < 64; ++d) {
        ap[d] = (_Float16)fsrc[(long long)d * NPTS];
        ax[d] = (_Float16)psrc[(long long)d * NPTS];
    }
    ap[64] = (_Float16)gx; ap[65] = (_Float16)gy; ap[66] = (_Float16)gz;
    ap[67] = (_Float16)qx; ap[68] = (_Float16)qy; ap[69] = (_Float16)qz;
    ax[64] = (_Float16)gx; ax[65] = (_Float16)gy; ax[66] = (_Float16)gz;
    ax[67] = (_Float16)qx; ax[68] = (_Float16)qy; ax[69] = (_Float16)qz;
#pragma unroll
    for (int d = 70; d < 96; ++d) { ap[d] = (_Float16)0.f; ax[d] = (_Float16)0.f; }
}

// ---------------------------------------------------------------------------
// 3) WMMA GEMM:  Y[M,N] = A[M,K]f16 (row-major) * W[N,K]f16^T.
//    Each wave computes a 16(M) x 64(N) block: one A fragment per k-step feeds
//    4 back-to-back v_wmma ops (4 accumulators).  Requires K%32==0, N%64==0.
//    Lane layouts used here (contiguous k order, same for A and B):
//      A: lane(l): M=l&15, v16h = {K=kh*16+0..15}, kh=l>>4
//      B: lane(l): N=l&15, v16h = {K=kh*16+0..15}
//      D: lane(l): N=l&15, acc[r] = row (r + 8*kh)
// ---------------------------------------------------------------------------
__global__ __launch_bounds__(256) void k_wmma_gemm_nt(
    const _Float16* __restrict__ A, const _Float16* __restrict__ W,
    float* __restrict__ Y, int M, int N, int Kd)
{
    const int lane = threadIdx.x & 31;
    const int wv   = threadIdx.x >> 5;
    const int nblks = N >> 6;                          // 64-column blocks
    const long long blk = (long long)blockIdx.x * 8 + wv;
    const long long totalBlks = (long long)(M >> 4) * nblks;
    if (blk >= totalBlks) return;
    const long long mt = blk / nblks;
    const int nb = (int)(blk % nblks);
    const int mr = lane & 15;     // A row / B col / D col within tile
    const int kh = lane >> 4;

    const _Float16* arow = A + (mt * 16 + mr) * (long long)Kd;
    const _Float16* w0   = W + (long long)(nb * 64 + mr) * Kd;
    const long long wstep = 16LL * Kd;                 // 16 columns of W

    v8f acc0 = {}, acc1 = {}, acc2 = {}, acc3 = {};
    for (int k0 = 0; k0 < Kd; k0 += 32) {
        const v16h a = *(const v16h*)(arow + k0 + kh * 16);
        const long long wo = (long long)k0 + kh * 16;
        const v16h b0 = *(const v16h*)(w0 + wo);
        const v16h b1 = *(const v16h*)(w0 + wstep + wo);
        const v16h b2 = *(const v16h*)(w0 + 2 * wstep + wo);
        const v16h b3 = *(const v16h*)(w0 + 3 * wstep + wo);
        acc0 = __builtin_amdgcn_wmma_f32_16x16x32_f16(false, a, false, b0, (short)0, acc0, false, false);
        acc1 = __builtin_amdgcn_wmma_f32_16x16x32_f16(false, a, false, b1, (short)0, acc1, false, false);
        acc2 = __builtin_amdgcn_wmma_f32_16x16x32_f16(false, a, false, b2, (short)0, acc2, false, false);
        acc3 = __builtin_amdgcn_wmma_f32_16x16x32_f16(false, a, false, b3, (short)0, acc3, false, false);
    }
    float* ybase = Y + (mt * 16 + 8 * kh) * (long long)N + nb * 64 + mr;
#pragma unroll
    for (int r = 0; r < 8; ++r) {
        float* yr = ybase + (long long)r * N;
        yr[0]  = acc0[r];
        yr[16] = acc1[r];
        yr[32] = acc2[r];
        yr[48] = acc3[r];
    }
}

// ---------------------------------------------------------------------------
// 4) Deterministic per-channel sum / sumsq for training-mode BN.
// ---------------------------------------------------------------------------
__global__ __launch_bounds__(256) void k_colstats_part(
    const float* __restrict__ Y, long long Mrows, int N, float* __restrict__ part)
{
    __shared__ float a1[256], a2[256];
    const int t = threadIdx.x;
    const int c = t % N, rsub = t / N, rstep = 256 / N;
    float s1 = 0.f, s2 = 0.f;
    for (long long r = (long long)blockIdx.x * rstep + rsub; r < Mrows;
         r += (long long)gridDim.x * rstep) {
        const float v = Y[r * N + c];
        s1 += v; s2 += v * v;
    }
    a1[t] = s1; a2[t] = s2;
    __syncthreads();
    if (rsub == 0) {
        for (int j = 1; j < rstep; ++j) { s1 += a1[j * N + c]; s2 += a2[j * N + c]; }
        part[(long long)blockIdx.x * 2 * N + c]     = s1;
        part[(long long)blockIdx.x * 2 * N + N + c] = s2;
    }
}

__global__ void k_colstats_final(const float* __restrict__ part, int nblocks, int N,
                                 float* __restrict__ st)
{
    const int c = blockIdx.x * 256 + threadIdx.x;
    if (c >= 2 * N) return;
    float s = 0.f;
    for (int b = 0; b < nblocks; ++b) s += part[(long long)b * 2 * N + c];
    st[c] = s;
}

// ---------------------------------------------------------------------------
// 5) BN + ReLU -> f16 (feeds second GEMM).
// ---------------------------------------------------------------------------
__global__ __launch_bounds__(256) void k_bnrelu_f16(
    const float* __restrict__ Y, const float* __restrict__ st,
    const float* __restrict__ g, const float* __restrict__ bt,
    _Float16* __restrict__ O, long long total, int N, float invM)
{
    const long long i = (long long)blockIdx.x * 256 + threadIdx.x;
    if (i >= total) return;
    const int c = (int)(i % N);
    const float mean = st[c] * invM;
    const float var  = st[N + c] * invM - mean * mean;
    const float sc   = g[c] * rsqrtf(var + 1e-5f);
    const float v    = sc * (Y[i] - mean) + bt[c];
    O[i] = (_Float16)(v > 0.f ? v : 0.f);
}

// ---------------------------------------------------------------------------
// 6) BN + ReLU fused with max-over-K neighbors -> F [BS, 128] (point-major).
//    ReLU folded into max (init 0).
// ---------------------------------------------------------------------------
__global__ __launch_bounds__(256) void k_kmax_bnrelu(
    const float* __restrict__ Y, const float* __restrict__ st,
    const float* __restrict__ g, const float* __restrict__ bt,
    float* __restrict__ F, float invM)
{
    const long long i = (long long)blockIdx.x * 256 + threadIdx.x;  // BS*128
    const int c = (int)(i & 127);
    const long long p = i >> 7;
    const float mean = st[c] * invM;
    const float var  = st[128 + c] * invM - mean * mean;
    const float sc   = g[c] * rsqrtf(var + 1e-5f);
    const float off  = bt[c] - mean * sc;
    const float* base = Y + p * NK * 128 + c;
    float m = 0.f;                       // ReLU floor
#pragma unroll 8
    for (int k = 0; k < NK; ++k) m = fmaxf(m, sc * base[k * 128] + off);
    F[i] = m;
}

// ---------------------------------------------------------------------------
// 7) Channel attention helpers.
// ---------------------------------------------------------------------------
__global__ void k_rowsum(const float* __restrict__ F, float* __restrict__ rs)
{
    const int p = blockIdx.x * 256 + threadIdx.x;
    if (p >= BS) return;
    const float* f = F + (long long)p * 128;
    float s = 0.f;
    for (int c = 0; c < 128; ++c) s += f[c];
    rs[p] = s;
}

__global__ void k_attsum(const float* __restrict__ F, const float* __restrict__ rs,
                         float* __restrict__ att)
{
    const int i = blockIdx.x * 256 + threadIdx.x;   // NB*128
    if (i >= NB * 128) return;
    const int b = i >> 7, c = i & 127;
    const float* f = F + (long long)b * NS * 128 + c;
    const float* r = rs + b * NS;
    float s = 0.f;
    for (int ss = 0; ss < NS; ++ss) s += f[(long long)ss * 128] * r[ss];
    att[i] = s;
}

// att [8,128] -> z = w1@att -> BN(over batch) -> ReLU -> sigmoid(w2@h) = scale [8,128]
__global__ __launch_bounds__(256) void k_att_mlp(
    const float* __restrict__ att, const float* __restrict__ w1,
    const float* __restrict__ g, const float* __restrict__ bb,
    const float* __restrict__ w2, float* __restrict__ scale)
{
    __shared__ float z[NB * 32], h[NB * 32], mo[32], vo[32];
    const int t = threadIdx.x;
    {
        const int bidx = t >> 5, o = t & 31;
        float s = 0.f;
        for (int c = 0; c < 128; ++c) s += w1[o * 128 + c] * att[bidx * 128 + c];
        z[bidx * 32 + o] = s;
    }
    __syncthreads();
    if (t < 32) {
        float s = 0.f, s2 = 0.f;
        for (int b = 0; b < NB; ++b) { const float v = z[b * 32 + t]; s += v; s2 += v * v; }
        const float m = s / NB;
        mo[t] = m; vo[t] = s2 / NB - m * m;
    }
    __syncthreads();
    {
        const int bidx = t >> 5, o = t & 31;
        const float r = g[o] * (z[bidx * 32 + o] - mo[o]) * rsqrtf(vo[o] + 1e-5f) + bb[o];
        h[bidx * 32 + o] = r > 0.f ? r : 0.f;
    }
    __syncthreads();
    for (int i = t; i < NB * 128; i += 256) {
        const int bidx = i >> 7, c = i & 127;
        float s = 0.f;
        for (int j = 0; j < 32; ++j) s += w2[c * 32 + j] * h[bidx * 32 + j];
        scale[i] = 1.f / (1.f + expf(-s));
    }
}

__global__ void k_scale_half(float* __restrict__ F, const float* __restrict__ sc,
                             _Float16* __restrict__ Fh)
{
    const int i = blockIdx.x * 256 + threadIdx.x;   // BS*128 = 1048576
    const int b = i >> 17, c = i & 127;
    const float v = F[i] * sc[b * 128 + c];
    F[i] = v; Fh[i] = (_Float16)v;
}

// ---------------------------------------------------------------------------
// 8) Weight convert + zero-pad to f16 [O, Cp].
// ---------------------------------------------------------------------------
__global__ void k_wpad(const float* __restrict__ w, _Float16* __restrict__ o,
                       int O, int C, int Cp)
{
    const int i = blockIdx.x * 256 + threadIdx.x;
    if (i >= O * Cp) return;
    const int r = i / Cp, c = i % Cp;
    o[i] = (c < C) ? (_Float16)w[r * C + c] : (_Float16)0.f;
}

// ---------------------------------------------------------------------------
// 9) CSA combine: rank-1 energy -> normalize -> softmax(channels) -> *V + residual.
//    One 128-thread block per point.
// ---------------------------------------------------------------------------
__global__ __launch_bounds__(128) void k_csa(
    const float* __restrict__ Q, const float* __restrict__ Km,
    const float* __restrict__ V, const float* __restrict__ Fin,
    float* __restrict__ Out)
{
    __shared__ float red[128];
    const int p = blockIdx.x, c = threadIdx.x;
    const long long o = (long long)p * 128 + c;
    const float q = Q[o], kk = Km[o], v = V[o];

    red[c] = q; __syncthreads();
    for (int off = 64; off > 0; off >>= 1) { if (c < off) red[c] += red[c + off]; __syncthreads(); }
    const float qs = red[0]; __syncthreads();

    const float e = qs * kk;
    red[c] = e; __syncthreads();
    for (int off = 64; off > 0; off >>= 1) { if (c < off) red[c] += red[c + off]; __syncthreads(); }
    const float es = red[0]; __syncthreads();

    const float nrm = e / (1e-9f + es);
    red[c] = nrm; __syncthreads();
    for (int off = 64; off > 0; off >>= 1) { if (c < off) red[c] = fmaxf(red[c], red[c + off]); __syncthreads(); }
    const float mx = red[0]; __syncthreads();

    const float w = expf(nrm - mx);
    red[c] = w; __syncthreads();
    for (int off = 64; off > 0; off >>= 1) { if (c < off) red[c] += red[c + off]; __syncthreads(); }
    const float sw = red[0];

    Out[o] = (w / sw) * v + Fin[o];
}

// ---------------------------------------------------------------------------
// 10) Assemble d_out = [new_xyz^T (B,3,S) | CF (B,128,S) | CP (B,128,S) | fps].
// ---------------------------------------------------------------------------
__global__ void k_finalize(const float* __restrict__ nxyz, const float* __restrict__ CF,
                           const float* __restrict__ CP, const int* __restrict__ fi,
                           float* __restrict__ out)
{
    const int i = blockIdx.x * 256 + threadIdx.x;
    const int A = NB * 3 * NS;          // 24576
    const int Bsz = NB * 128 * NS;      // 1048576
    const int total = A + 2 * Bsz + BS;
    if (i >= total) return;
    if (i < A) {
        const int b = i / 3072, r = i % 3072, c = r / NS, s = r % NS;
        out[i] = nxyz[((long long)b * NS + s) * 3 + c];
    } else if (i < A + Bsz) {
        const int j = i - A; const int b = j >> 17, r = j & 131071, c = r >> 10, s = r & 1023;
        out[i] = CF[(((long long)b << 10) + s) * 128 + c];
    } else if (i < A + 2 * Bsz) {
        const int j = i - A - Bsz; const int b = j >> 17, r = j & 131071, c = r >> 10, s = r & 1023;
        out[i] = CP[(((long long)b << 10) + s) * 128 + c];
    } else {
        out[i] = (float)fi[i - A - 2 * Bsz];
    }
}

// ---------------------------------------------------------------------------
// Host orchestration (graph-capture safe: stream-only, scratch from d_ws).
// ---------------------------------------------------------------------------
extern "C" void kernel_launch(void* const* d_in, const int* in_sizes, int n_in,
                              void* d_out, int out_size, void* d_ws, size_t ws_size,
                              hipStream_t stream)
{
    (void)in_sizes; (void)n_in; (void)out_size; (void)ws_size;
    const float* xyz  = (const float*)d_in[0];
    const float* csaf = (const float*)d_in[1];
    const float* csap = (const float*)d_in[2];
    const float *fw1=(const float*)d_in[3],  *fg1=(const float*)d_in[4],  *fb1=(const float*)d_in[5];
    const float *fw2=(const float*)d_in[6],  *fg2=(const float*)d_in[7],  *fb2=(const float*)d_in[8];
    const float *pw1=(const float*)d_in[9],  *pg1=(const float*)d_in[10], *pb1=(const float*)d_in[11];
    const float *pw2=(const float*)d_in[12], *pg2=(const float*)d_in[13], *pb2=(const float*)d_in[14];
    const float *a1w1=(const float*)d_in[15],*a1g=(const float*)d_in[16], *a1b=(const float*)d_in[17], *a1w2=(const float*)d_in[18];
    const float *a2w1=(const float*)d_in[19],*a2g=(const float*)d_in[20], *a2b=(const float*)d_in[21], *a2w2=(const float*)d_in[22];
    const float *cfq=(const float*)d_in[23], *cfk=(const float*)d_in[24], *cfv=(const float*)d_in[25];
    const float *cpq=(const float*)d_in[26], *cpk=(const float*)d_in[27], *cpv=(const float*)d_in[28];

    char* wsb = (char*)d_ws;
    size_t off = 0;
    auto carve = [&](size_t bytes) -> char* {
        char* p = wsb + off;
        off = (off + bytes + 255) & ~(size_t)255;
        return p;
    };
    float*     nxyz  = (float*)carve(BS * 3 * 4);
    int*       fidx  = (int*)carve(BS * 4);
    _Float16*  Apts  = (_Float16*)carve((size_t)BSK * 96 * 2);
    _Float16*  Axyz  = (_Float16*)carve((size_t)BSK * 96 * 2);
    float*     Ybuf  = (float*)carve((size_t)BSK * 128 * 4);   // Y1 (first half) then Y2
    _Float16*  X2    = (_Float16*)carve((size_t)BSK * 64 * 2);
    float*     part  = (float*)carve((size_t)1024 * 256 * 4);
    float*     stats = (float*)carve(256 * 4);
    float*     Fbuf  = (float*)carve((size_t)BS * 128 * 4);
    float*     Pbuf  = (float*)carve((size_t)BS * 128 * 4);
    _Float16*  Fh    = (_Float16*)carve((size_t)BS * 128 * 2);
    _Float16*  Ph    = (_Float16*)carve((size_t)BS * 128 * 2);
    float*     Qb    = (float*)carve((size_t)BS * 128 * 4);
    float*     Kb    = (float*)carve((size_t)BS * 128 * 4);
    float*     Vb    = (float*)carve((size_t)BS * 128 * 4);
    float*     CF    = (float*)carve((size_t)BS * 128 * 4);
    float*     CP    = (float*)carve((size_t)BS * 128 * 4);
    float*     rsb   = (float*)carve(BS * 4);
    float*     attb  = (float*)carve(NB * 128 * 4);
    float*     scb   = (float*)carve(NB * 128 * 4);
    _Float16*  w1h   = (_Float16*)carve(64 * 96 * 2);
    _Float16*  w2h   = (_Float16*)carve(128 * 64 * 2);
    _Float16*  wch   = (_Float16*)carve(128 * 128 * 2);

    k_fps<<<NB, 256, 0, stream>>>(xyz, nxyz, fidx);
    k_knn_gather<<<NB * 128, 256, 0, stream>>>(xyz, csaf, csap, nxyz, Apts, Axyz);

    auto gemm = [&](const _Float16* A, const _Float16* W, float* Y, int M, int N, int Kd) {
        const long long blks = (long long)(M / 16) * (N / 64);   // 16x64 per wave
        k_wmma_gemm_nt<<<(int)((blks + 7) / 8), 256, 0, stream>>>(A, W, Y, M, N, Kd);
    };
    auto wpad = [&](const float* w, _Float16* o, int O, int C, int Cp) {
        const int n = O * Cp;
        k_wpad<<<(n + 255) / 256, 256, 0, stream>>>(w, o, O, C, Cp);
    };
    auto colstats = [&](const float* Y, long long M, int N) {
        k_colstats_part<<<1024, 256, 0, stream>>>(Y, M, N, part);
        k_colstats_final<<<1, 256, 0, stream>>>(part, 1024, N, stats);
    };

    const long long Mrows = BSK;
    const float invM = 1.0f / (float)Mrows;

    auto run_path = [&](const _Float16* Ain,
                        const float* w1, const float* g1, const float* b1,
                        const float* w2, const float* g2, const float* b2,
                        const float* aw1, const float* ag, const float* ab, const float* aw2,
                        float* Fout, _Float16* Fhout) {
        wpad(w1, w1h, 64, 70, 96);
        wpad(w2, w2h, 128, 64, 64);
        gemm(Ain, w1h, Ybuf, (int)Mrows, 64, 96);                       // v_wmma f16
        colstats(Ybuf, Mrows, 64);
        {
            const long long tot = Mrows * 64;
            k_bnrelu_f16<<<(int)((tot + 255) / 256), 256, 0, stream>>>(
                Ybuf, stats, g1, b1, X2, tot, 64, invM);
        }
        gemm(X2, w2h, Ybuf, (int)Mrows, 128, 64);                       // v_wmma f16
        colstats(Ybuf, Mrows, 128);
        k_kmax_bnrelu<<<(int)(BS * 128 / 256), 256, 0, stream>>>(Ybuf, stats, g2, b2, Fout, invM);
        k_rowsum<<<BS / 256, 256, 0, stream>>>(Fout, rsb);
        k_attsum<<<(NB * 128 + 255) / 256, 256, 0, stream>>>(Fout, rsb, attb);
        k_att_mlp<<<1, 256, 0, stream>>>(attb, aw1, ag, ab, aw2, scb);
        k_scale_half<<<(int)(BS * 128 / 256), 256, 0, stream>>>(Fout, scb, Fhout);
    };

    // feature path uses a2*, position path uses a1* (per reference)
    run_path(Apts, fw1, fg1, fb1, fw2, fg2, fb2, a2w1, a2g, a2b, a2w2, Fbuf, Fh);
    run_path(Axyz, pw1, pg1, pb1, pw2, pg2, pb2, a1w1, a1g, a1b, a1w2, Pbuf, Ph);

    // csa_feature: q,k from position; v from feature; residual = feature
    wpad(cfq, wch, 128, 128, 128); gemm(Ph, wch, Qb, BS, 128, 128);
    wpad(cfk, wch, 128, 128, 128); gemm(Ph, wch, Kb, BS, 128, 128);
    wpad(cfv, wch, 128, 128, 128); gemm(Fh, wch, Vb, BS, 128, 128);
    k_csa<<<BS, 128, 0, stream>>>(Qb, Kb, Vb, Fbuf, CF);

    // csa_position: q,k from feature; v from position; residual = position
    wpad(cpq, wch, 128, 128, 128); gemm(Fh, wch, Qb, BS, 128, 128);
    wpad(cpk, wch, 128, 128, 128); gemm(Fh, wch, Kb, BS, 128, 128);
    wpad(cpv, wch, 128, 128, 128); gemm(Ph, wch, Vb, BS, 128, 128);
    k_csa<<<BS, 128, 0, stream>>>(Qb, Kb, Vb, Pbuf, CP);

    const int total = NB * 3 * NS + 2 * NB * 128 * NS + BS;   // 2129920
    k_finalize<<<(total + 255) / 256, 256, 0, stream>>>(nxyz, CF, CP, fidx, (float*)d_out);
}
